// GraphNetwork_76570676953656
// MI455X (gfx1250) — matrix operand order repaired and checked by reading, not compiled, not run on hardware
//
#include <hip/hip_runtime.h>

#define NN   16384
#define DIM  32
#define OUTD 16
#define BN_EPS 1e-5f
#define LROW  40   // padded row (halves) for 32-wide f16 tiles: 80 B
#define LROWA 72   // padded row (halves) for 64-wide adj tile: 144 B (16B-aligned frags)

typedef __attribute__((ext_vector_type(16))) _Float16 v16h;
typedef __attribute__((ext_vector_type(8)))  _Float16 v8h;
typedef __attribute__((ext_vector_type(8)))  float    v8f;

__device__ __forceinline__ v16h cat8(v8h a, v8h b) {
  return __builtin_shufflevector(a, b, 0,1,2,3,4,5,6,7,8,9,10,11,12,13,14,15);
}

__device__ __forceinline__ _Float16 adj2h(int v) {
  return v ? (_Float16)1.0f : (_Float16)0.0f;
}

// ---------------------------------------------------------------------------
// Prep: xT[f][j] = (f16) x[j][f]   (32 x NN, 1 MB, lives in L2)
// ---------------------------------------------------------------------------
__global__ __launch_bounds__(128) void k_prep_xT(const float* __restrict__ x,
                                                 _Float16* __restrict__ xT)
{
  __shared__ _Float16 sh[32 * 33];
  const int tid = threadIdx.x;
  const int j0  = blockIdx.x * 32;
#pragma unroll
  for (int r = 0; r < 2; ++r) {
    int q = tid + r * 128;
    int m = q >> 3, g = q & 7;
    float4 v = *(const float4*)&x[(size_t)(j0 + m) * DIM + g * 4];
    _Float16* p = &sh[(g * 4) * 33 + m];
    p[0]  = (_Float16)v.x;
    p[33] = (_Float16)v.y;
    p[66] = (_Float16)v.z;
    p[99] = (_Float16)v.w;
  }
  __syncthreads();
#pragma unroll
  for (int r = 0; r < 8; ++r) {
    int q = tid + r * 128;            // f = q>>5, jj = q&31  (coalesced stores)
    int f = q >> 5, jj = q & 31;
    xT[(size_t)f * NN + j0 + jj] = sh[f * 33 + jj];
  }
}

// ---------------------------------------------------------------------------
// Kernel 1: agg = adj^T @ x (WMMA f16, K-step 64), h = x + agg,
//           hpre = h@W1 + b1 -> ws, BN sum/sumsq atomics.
// ---------------------------------------------------------------------------
__global__ __launch_bounds__(128) void k1_gin_agg_mlp1(
    const float* __restrict__ x, const int* __restrict__ adj,
    const _Float16* __restrict__ xT,
    const float* __restrict__ W1, const float* __restrict__ b1,
    float* __restrict__ h1, float* __restrict__ bn_red)
{
  __shared__ __align__(16) _Float16 lds_a  [32 * LROWA]; // adjT tile [m][k], k<64
  __shared__ __align__(16) _Float16 lds_h  [32 * LROW];  // h tile    [m][k]
  __shared__ __align__(16) _Float16 lds_w1T[32 * LROW];  // W1        [n][k]
  __shared__ float lds_xd[32 * 32];
  __shared__ float red[64];

  const int tid = threadIdx.x;
  const int i0  = blockIdx.x * 32;

  for (int q = tid; q < 1024; q += 128) {
    int k = q >> 5, n = q & 31;                  // W1 row-major [k][n]
    lds_w1T[n * LROW + k] = (_Float16)W1[q];
    lds_xd[q] = x[(size_t)i0 * DIM + q];
  }
  if (tid < 64) red[tid] = 0.f;

  const int wv    = tid >> 5;
  const int lane  = tid & 31;
  const int mhalf = wv >> 1;
  const int l16   = lane & 15;
  const int lhi   = lane >> 4;
  const int arow  = mhalf * 16 + l16;
  const int nrow  = (wv & 1) * 16 + l16;
  const int kbase = lhi * 8;

  // Per-lane B row pointer into xT (already offset by half-wave K base).
  const _Float16* xrow = xT + (size_t)nrow * NN + lhi * 8;

  v8f acc = {};

  for (int j0 = 0; j0 < NN; j0 += 64) {
    int4 av[4];
    int  kk[4], gg[4];
#pragma unroll
    for (int r = 0; r < 4; ++r) {
      int q = tid + r * 128;          // [0,512): 64 source rows x 8 int4 groups
      int k = q >> 3, g = q & 7;
      kk[r] = k; gg[r] = g;
      av[r] = *(const int4*)&adj[(size_t)(j0 + k) * NN + i0 + g * 4];
    }
    {   // prefetch next adj chunk (speculative, dropped if redundant)
      int jn = (j0 + 64 < NN) ? (j0 + 64) : j0;
      __builtin_prefetch(&adj[(size_t)(jn + (tid >> 3)) * NN + i0 + (tid & 7) * 4], 0, 3);
    }
    __syncthreads();                  // prior iteration's A-fragment reads done
#pragma unroll
    for (int r = 0; r < 4; ++r) {
      _Float16* pa = &lds_a[(gg[r] * 4) * LROWA + kk[r]];   // transpose [m][k]
      pa[0 * LROWA] = adj2h(av[r].x);
      pa[1 * LROWA] = adj2h(av[r].y);
      pa[2 * LROWA] = adj2h(av[r].z);
      pa[3 * LROWA] = adj2h(av[r].w);
    }
    __syncthreads();
    v16h B0 = cat8(*(const v8h*)&xrow[j0],      *(const v8h*)&xrow[j0 + 16]);
    v16h B1 = cat8(*(const v8h*)&xrow[j0 + 32], *(const v8h*)&xrow[j0 + 48]);
    v16h A0 = cat8(*(const v8h*)&lds_a[arow * LROWA + kbase],
                   *(const v8h*)&lds_a[arow * LROWA + 16 + kbase]);
    v16h A1 = cat8(*(const v8h*)&lds_a[arow * LROWA + 32 + kbase],
                   *(const v8h*)&lds_a[arow * LROWA + 48 + kbase]);
    acc = __builtin_amdgcn_wmma_f32_16x16x32_f16(false, A0, false, B0,
                                                 (short)0, acc, false, false);
    acc = __builtin_amdgcn_wmma_f32_16x16x32_f16(false, A1, false, B1,
                                                 (short)0, acc, false, false);
  }

  // h = x + agg -> lds_h (f16); C/D layout: VGPR r holds M = r + 8*lhi.
  __syncthreads();
#pragma unroll
  for (int r = 0; r < 8; ++r) {
    int m = mhalf * 16 + lhi * 8 + r;
    float v = acc[r] + lds_xd[m * 32 + nrow];
    lds_h[m * LROW + nrow] = (_Float16)v;
  }
  __syncthreads();

  // hpre = h @ W1 + b1
  v16h A2 = cat8(*(const v8h*)&lds_h  [arow * LROW + kbase],
                 *(const v8h*)&lds_h  [arow * LROW + 16 + kbase]);
  v16h B2 = cat8(*(const v8h*)&lds_w1T[nrow * LROW + kbase],
                 *(const v8h*)&lds_w1T[nrow * LROW + 16 + kbase]);
  v8f acc2 = {};
  acc2 = __builtin_amdgcn_wmma_f32_16x16x32_f16(false, A2, false, B2,
                                                (short)0, acc2, false, false);

  const float bias = b1[nrow];
  float s = 0.f, s2 = 0.f;
#pragma unroll
  for (int r = 0; r < 8; ++r) {
    int m = mhalf * 16 + lhi * 8 + r;
    float v = acc2[r] + bias;
    h1[(size_t)(i0 + m) * DIM + nrow] = v;
    s  += v;
    s2 += v * v;
  }
  atomicAdd(&red[nrow],      s);
  atomicAdd(&red[32 + nrow], s2);
  __syncthreads();
  if (tid < 32) {
    atomicAdd(&bn_red[tid],      red[tid]);
    atomicAdd(&bn_red[32 + tid], red[32 + tid]);
  }
}

// ---------------------------------------------------------------------------
// Kernel 2: y = relu(BN(hpre)); h2 = relu(y @ W2 + b2); pool += colsum(h2)
// ---------------------------------------------------------------------------
__global__ __launch_bounds__(128) void k2_bn_mlp2_pool(
    const float* __restrict__ h1, const float* __restrict__ bn_red,
    const float* __restrict__ gamma, const float* __restrict__ beta,
    const float* __restrict__ W2, const float* __restrict__ b2,
    float* __restrict__ pool)
{
  __shared__ __align__(16) _Float16 lds_h  [32 * LROW];
  __shared__ __align__(16) _Float16 lds_w2T[32 * LROW];
  __shared__ float s_scale[32], s_shift[32];
  __shared__ float red[32];

  const int tid = threadIdx.x;
  const int i0  = blockIdx.x * 32;

  for (int q = tid; q < 1024; q += 128) {
    int k = q >> 5, n = q & 31;
    lds_w2T[n * LROW + k] = (_Float16)W2[q];
  }
  if (tid < 32) {
    float mu  = bn_red[tid] * (1.0f / NN);
    float var = bn_red[32 + tid] * (1.0f / NN) - mu * mu;
    float sc  = gamma[tid] * rsqrtf(var + BN_EPS);
    s_scale[tid] = sc;
    s_shift[tid] = beta[tid] - mu * sc;
    red[tid] = 0.f;
  }
  __syncthreads();

#pragma unroll
  for (int r = 0; r < 2; ++r) {
    int q = tid + r * 128;
    int m = q >> 3, g = q & 7;
    float4 v = *(const float4*)&h1[(size_t)(i0 + m) * DIM + g * 4];
    _Float16* ph = &lds_h[m * LROW + g * 4];
    ph[0] = (_Float16)fmaxf(v.x * s_scale[g * 4 + 0] + s_shift[g * 4 + 0], 0.f);
    ph[1] = (_Float16)fmaxf(v.y * s_scale[g * 4 + 1] + s_shift[g * 4 + 1], 0.f);
    ph[2] = (_Float16)fmaxf(v.z * s_scale[g * 4 + 2] + s_shift[g * 4 + 2], 0.f);
    ph[3] = (_Float16)fmaxf(v.w * s_scale[g * 4 + 3] + s_shift[g * 4 + 3], 0.f);
  }
  __syncthreads();

  const int wv    = tid >> 5;
  const int lane  = tid & 31;
  const int mhalf = wv >> 1;
  const int l16   = lane & 15;
  const int lhi   = lane >> 4;
  const int arow  = mhalf * 16 + l16;
  const int nrow  = (wv & 1) * 16 + l16;
  const int kbase = lhi * 8;

  v16h A = cat8(*(const v8h*)&lds_h  [arow * LROW + kbase],
                *(const v8h*)&lds_h  [arow * LROW + 16 + kbase]);
  v16h B = cat8(*(const v8h*)&lds_w2T[nrow * LROW + kbase],
                *(const v8h*)&lds_w2T[nrow * LROW + 16 + kbase]);
  v8f acc = {};
  acc = __builtin_amdgcn_wmma_f32_16x16x32_f16(false, A, false, B,
                                               (short)0, acc, false, false);

  const float bias = b2[nrow];
  float s = 0.f;
#pragma unroll
  for (int r = 0; r < 8; ++r) {
    s += fmaxf(acc[r] + bias, 0.f);
  }
  atomicAdd(&red[nrow], s);
  __syncthreads();
  if (tid < 32) atomicAdd(&pool[tid], red[tid]);
}

// ---------------------------------------------------------------------------
// Kernel 3: out = (pool/N) @ Wf + bf   (1x32 @ 32x16)
// ---------------------------------------------------------------------------
__global__ void k3_fc(const float* __restrict__ pool, const float* __restrict__ Wf,
                      const float* __restrict__ bf, float* __restrict__ out)
{
  int o = threadIdx.x;
  if (o < OUTD) {
    float acc = bf[o];
    for (int k = 0; k < DIM; ++k)
      acc += (pool[k] * (1.0f / NN)) * Wf[k * OUTD + o];
    out[o] = acc;
  }
}

__global__ void k0_zero(float* __restrict__ p, int n)
{
  int t = blockIdx.x * blockDim.x + threadIdx.x;
  if (t < n) p[t] = 0.f;
}

// ---------------------------------------------------------------------------
extern "C" void kernel_launch(void* const* d_in, const int* in_sizes, int n_in,
                              void* d_out, int out_size, void* d_ws, size_t ws_size,
                              hipStream_t stream)
{
  (void)in_sizes; (void)n_in; (void)out_size; (void)ws_size;
  const float* x     = (const float*)d_in[0];
  const int*   adj   = (const int*)  d_in[1];
  const float* W1    = (const float*)d_in[2];
  const float* b1    = (const float*)d_in[3];
  const float* gamma = (const float*)d_in[4];
  const float* beta  = (const float*)d_in[5];
  const float* W2    = (const float*)d_in[6];
  const float* b2    = (const float*)d_in[7];
  const float* Wf    = (const float*)d_in[8];
  const float* bf    = (const float*)d_in[9];
  float* out = (float*)d_out;

  float*     h1     = (float*)d_ws;                    // N*32 f32 (2 MB)
  float*     bn_red = h1 + (size_t)NN * DIM;           // 64 f32: sum / sumsq
  float*     pool   = bn_red + 64;                     // 32 f32
  _Float16*  xT     = (_Float16*)(pool + 32);          // 32 x NN f16 (1 MB), 16B-aligned

  k0_zero        <<<1,       128, 0, stream>>>(bn_red, 96);
  k_prep_xT      <<<NN / 32, 128, 0, stream>>>(x, xT);
  k1_gin_agg_mlp1<<<NN / 32, 128, 0, stream>>>(x, adj, xT, W1, b1, h1, bn_red);
  k2_bn_mlp2_pool<<<NN / 32, 128, 0, stream>>>(h1, bn_red, gamma, beta, W2, b2, pool);
  k3_fc          <<<1,        32, 0, stream>>>(pool, Wf, bf, out);
}
